// CombinedLoss_7086696038745
// MI455X (gfx1250) — matrix operand order repaired and checked by reading, not compiled, not run on hardware
//
#include <hip/hip_runtime.h>
#include <math.h>

// -------- CDNA5 (gfx1250) WMMA types --------
typedef _Float16 f16;
typedef __attribute__((ext_vector_type(16))) _Float16 v16h;
typedef __attribute__((ext_vector_type(8)))  float    v8f;

#define IMG_H 512
#define IMG_W 512
#define TILE  16
#define TILES_X (IMG_W / TILE)   // 32
#define TILES_Y (IMG_H / TILE)   // 32
#define WAVES_PER_WG 4
#define WG_SIZE (WAVES_PER_WG * 32)

// Gaussian(sigma=1.5) 11-tap, normalized. 1/sum(exp(-(i-5)^2/4.5)) precomputed.
__device__ __forceinline__ float gauss_w(int d) {
    float t = (float)(d - 5);
    return __expf(-t * t * (1.0f / 4.5f)) * (1.0f / 3.7592308f);
}

// Each wave computes one 16x16 output tile of the SSIM map + its L1 partial.
// Conv is separable: horizontal then vertical, each as a 16x16x32 f16 WMMA
// against a constant banded Gaussian matrix (K=32 covers 16 outputs + 10 taps).
__global__ __launch_bounds__(WG_SIZE) void ssim_l1_main(
    const float* __restrict__ img1, const float* __restrict__ img2,
    float* __restrict__ partial)
{
    // Per-wave LDS slices (no cross-wave sharing until final reduce).
    __shared__ f16 s_ch[WAVES_PER_WG][5][32][32];   // product channels, row-major
    __shared__ f16 s_hv[WAVES_PER_WG][5][16][32];   // Hconv, column-major [col][row]
    __shared__ float s_r0[WG_SIZE];
    __shared__ float s_r1[WG_SIZE];

    const int tid  = threadIdx.x;
    const int wv   = tid >> 5;
    const int lane = tid & 31;
    const int hf   = lane >> 4;   // lane half: 0 or 1
    const int l15  = lane & 15;

    // ---- constant banded Gaussian matrices (per-lane WMMA layout) ----
    // B_h (32x16): K = (e>>3)*16 + 8*hf + (e&7), N = l15, val = w[K - N] if 0<=K-N<=10
    // A_v (16x32): M = l15, K = (e>>3)*16 + 8*hf + (e&7), val = w[K - M]
    v16h Bh, Av;
#pragma unroll
    for (int e = 0; e < 16; ++e) {
        int kb = ((e >> 3) * 16) + hf * 8 + (e & 7);
        int db = kb - l15;
        Bh[e] = (db >= 0 && db <= 10) ? (f16)gauss_w(db) : (f16)0.0f;
        int ka = ((e >> 3) * 16) + hf * 8 + (e & 7);
        int da = ka - l15;
        Av[e] = (da >= 0 && da <= 10) ? (f16)gauss_w(da) : (f16)0.0f;
    }

    // ---- tile coordinates ----
    const int tileId = blockIdx.x * WAVES_PER_WG + wv;
    const int img    = tileId / (TILES_X * TILES_Y);
    const int rem    = tileId % (TILES_X * TILES_Y);
    const int Y0     = (rem / TILES_X) * TILE;
    const int X0     = (rem % TILES_X) * TILE;
    const float* p1  = img1 + (size_t)img * IMG_H * IMG_W;
    const float* p2  = img2 + (size_t)img * IMG_H * IMG_W;

    // ---- stage 0: halo load + 5 product channels (f16) + L1 partial ----
    float l1_acc = 0.0f;
#pragma unroll
    for (int k = 0; k < 32; ++k) {           // 32x32 region, one row per iter
        int r = k, c = lane;
        float x1 = 0.0f, x2 = 0.0f;
        int gy = Y0 - 5 + r, gx = X0 - 5 + c;
        if (r < 26 && c < 26 && gy >= 0 && gy < IMG_H && gx >= 0 && gx < IMG_W) {
            x1 = p1[gy * IMG_W + gx];
            x2 = p2[gy * IMG_W + gx];
        }
        s_ch[wv][0][r][c] = (f16)x1;
        s_ch[wv][1][r][c] = (f16)x2;
        s_ch[wv][2][r][c] = (f16)(x1 * x1);
        s_ch[wv][3][r][c] = (f16)(x2 * x2);
        s_ch[wv][4][r][c] = (f16)(x1 * x2);
        if (r >= 5 && r < 21 && c >= 5 && c < 21) l1_acc += fabsf(x1 - x2);
    }
    __builtin_amdgcn_wave_barrier();   // order LDS stores vs. cross-lane gathers

    const v8f zeroC = {};

    // ---- stage 1: horizontal conv via WMMA (A = data, B = weights) ----
#pragma unroll
    for (int ch = 0; ch < 5; ++ch) {
#pragma unroll
        for (int g = 0; g < 2; ++g) {
            v16h a;
            const int row = g * 16 + l15;         // A: M = l15 within group
#pragma unroll
            for (int e = 0; e < 16; ++e) {
                int K = ((e >> 3) * 16) + hf * 8 + (e & 7);
                a[e] = s_ch[wv][ch][row][K];
            }
            v8f dh = __builtin_amdgcn_wmma_f32_16x16x32_f16(
                false, a, false, Bh, (short)0, zeroC, false, false);
            // D: M' = r + 8*hf (within group), N = l15; store column-major
#pragma unroll
            for (int r = 0; r < 8; ++r) {
                int rr = g * 16 + hf * 8 + r;
                s_hv[wv][ch][l15][rr] = (f16)dh[r];
            }
        }
    }
    __builtin_amdgcn_wave_barrier();

    // ---- stage 2: vertical conv via WMMA (A = weights, B = Hconv) ----
    v8f acc[5];
#pragma unroll
    for (int ch = 0; ch < 5; ++ch) {
        v16h b;                                   // B: K = (e>>3)*16 + 8*hf + (e&7), N = l15
#pragma unroll
        for (int e = 0; e < 16; ++e) {
            int K = ((e >> 3) * 16) + hf * 8 + (e & 7);
            b[e] = s_hv[wv][ch][l15][K];
        }
        acc[ch] = __builtin_amdgcn_wmma_f32_16x16x32_f16(
            false, Av, false, b, (short)0, zeroC, false, false);
    }

    // ---- stage 3: SSIM map (8 pixels per lane: M = r + 8*hf, N = l15) ----
    const float C1 = 1.0e-4f;   // (0.01*1.0)^2
    const float C2 = 9.0e-4f;   // (0.03*1.0)^2
    float ssim_acc = 0.0f;
#pragma unroll
    for (int r = 0; r < 8; ++r) {
        float mu1 = acc[0][r], mu2 = acc[1][r];
        float e11 = acc[2][r], e22 = acc[3][r], e12 = acc[4][r];
        float m1s = mu1 * mu1, m2s = mu2 * mu2, m12 = mu1 * mu2;
        float s1  = e11 - m1s, s2 = e22 - m2s, s12 = e12 - m12;
        float num = (2.0f * m12 + C1) * (2.0f * s12 + C2);
        float den = (m1s + m2s + C1) * (s1 + s2 + C2);
        ssim_acc += num / den;
    }

    // ---- deterministic block reduce -> per-block partial ----
    s_r0[tid] = ssim_acc;
    s_r1[tid] = l1_acc;
    __syncthreads();
#pragma unroll
    for (int s = WG_SIZE / 2; s > 0; s >>= 1) {
        if (tid < s) { s_r0[tid] += s_r0[tid + s]; s_r1[tid] += s_r1[tid + s]; }
        __syncthreads();
    }
    if (tid == 0) {
        partial[2 * blockIdx.x]     = s_r0[0];
        partial[2 * blockIdx.x + 1] = s_r1[0];
    }
}

// Deterministic final reduction: fixed-order strided sums + LDS tree.
__global__ __launch_bounds__(256) void ssim_l1_finalize(
    const float* __restrict__ partial, int nblocks,
    float* __restrict__ out, float inv_npx)
{
    __shared__ float sa[256];
    __shared__ float sb[256];
    int t = threadIdx.x;
    float a = 0.0f, b = 0.0f;
    for (int i = t; i < nblocks; i += 256) {
        a += partial[2 * i];
        b += partial[2 * i + 1];
    }
    sa[t] = a; sb[t] = b;
    __syncthreads();
#pragma unroll
    for (int s = 128; s > 0; s >>= 1) {
        if (t < s) { sa[t] += sa[t + s]; sb[t] += sb[t + s]; }
        __syncthreads();
    }
    if (t == 0) {
        float ssim = sa[0] * inv_npx;
        ssim = fminf(fmaxf(ssim, 0.0f), 1.0f);
        float l1 = sb[0] * inv_npx;
        out[0] = 0.5f * (1.0f - ssim) + 0.5f * l1;
    }
}

extern "C" void kernel_launch(void* const* d_in, const int* in_sizes, int n_in,
                              void* d_out, int out_size, void* d_ws, size_t ws_size,
                              hipStream_t stream) {
    const float* img1 = (const float*)d_in[0];
    const float* img2 = (const float*)d_in[1];
    float* out     = (float*)d_out;
    float* partial = (float*)d_ws;

    const long long total  = (long long)in_sizes[0];            // N*1*512*512
    const int       images = (int)(total / (IMG_H * IMG_W));    // 32
    const int       tiles  = images * TILES_X * TILES_Y;        // 32768
    const int       blocks = tiles / WAVES_PER_WG;              // 8192

    ssim_l1_main<<<blocks, WG_SIZE, 0, stream>>>(img1, img2, partial);

    const float inv_npx = 1.0f / (float)total;
    ssim_l1_finalize<<<1, 256, 0, stream>>>(partial, blocks, out, inv_npx);
}
